// AdvancedMultiHeadAttention_18270790877765
// MI455X (gfx1250) — matrix operand order repaired and checked by reading, not compiled, not run on hardware
//
#include <hip/hip_runtime.h>
#include <hip/hip_bf16.h>
#include <math.h>

// ---------------- problem constants (match reference setup_inputs) ----------
static constexpr int BB = 2, TT = 2048, CC = 1024, HH = 16, DD = 64;
static constexpr int MM = BB * TT;           // 4096 rows total

// ---------------- WMMA types / helpers --------------------------------------
typedef _Float16 f16x16 __attribute__((ext_vector_type(16)));
typedef float    f32x8  __attribute__((ext_vector_type(8)));

__device__ __forceinline__ unsigned short f2h(float f) {
  _Float16 h = (_Float16)f;
  return __builtin_bit_cast(unsigned short, h);
}
__device__ __forceinline__ float h2f(unsigned short u) {
  _Float16 h = __builtin_bit_cast(_Float16, u);
  return (float)h;
}

union Frag { f16x16 v; uint4 q[2]; };

// A fragment: 16(M) x 32(K) f16, from row-major [m][k] with row stride ld.
// lane m = lane&15 ; half = lane>>4 ; elems e -> K = half*16 + e (contiguous 16 halves).
__device__ __forceinline__ f16x16 load_frag_A(const unsigned short* base, int ld, int lane) {
  const int half = lane >> 4, m = lane & 15;
  Frag f;
  const unsigned short* p = base + (size_t)m * ld + half * 16;
  f.q[0] = *(const uint4*)(p);
  f.q[1] = *(const uint4*)(p + 8);
  return f.v;
}
// B fragment: 32(K) x 16(N) f16, from "N-major" storage [n][k] with row stride ld.
// lane n = lane&15 ; elems e -> K = half*16 + e (contiguous 16 halves).
__device__ __forceinline__ f16x16 load_frag_B(const unsigned short* base, int ld, int lane) {
  const int half = lane >> 4, n = lane & 15;
  Frag f;
  const unsigned short* p = base + (size_t)n * ld + half * 16;
  f.q[0] = *(const uint4*)(p);
  f.q[1] = *(const uint4*)(p + 8);
  return f.v;
}
__device__ __forceinline__ f32x8 wmma_f16(f16x16 a, f16x16 b, f32x8 c) {
  return __builtin_amdgcn_wmma_f32_16x16x32_f16(false, a, false, b, (short)0, c, false, false);
}

// ---------------- elementwise kernels ---------------------------------------
__global__ __launch_bounds__(256) void f32_to_f16_kernel(const float* __restrict__ src,
                                                         unsigned short* __restrict__ dst, int n) {
  int i = blockIdx.x * blockDim.x + threadIdx.x;
  if (i < n) dst[i] = f2h(src[i]);
}

// Read projected q/k (f16, [b][t][h*D+d]) -> apply RoPE -> write [b][h][t][d].
// scale folds the 1/sqrt(D) attention scale into q.
__global__ __launch_bounds__(256) void rope_split_kernel(const unsigned short* __restrict__ src,
                                                         unsigned short* __restrict__ dst,
                                                         float scale) {
  const int HD2 = DD / 2;
  int idx = blockIdx.x * blockDim.x + threadIdx.x;
  if (idx >= BB * TT * HH * HD2) return;
  int dp = idx % HD2; int r = idx / HD2;
  int h  = r % HH;    r /= HH;
  int t  = r % TT;    int b = r / TT;
  float inv_freq = expf(-((float)(2 * dp) / (float)DD) * 9.210340371976184f); // ln(1e4)
  float ang = (float)t * inv_freq;
  float cs = cosf(ang), sn = sinf(ang);
  const unsigned short* sp = src + (size_t)(b * TT + t) * CC + h * DD;
  float x1 = h2f(sp[2 * dp]), x2 = h2f(sp[2 * dp + 1]);
  unsigned short* dq = dst + ((size_t)(b * HH + h) * TT + t) * DD;
  dq[dp]       = f2h((x1 * cs - x2 * sn) * scale);
  dq[dp + HD2] = f2h((x1 * sn + x2 * cs) * scale);
}

// v (f16, [b][t][h*D+d]) -> vt [b][h][d][t]  (so P*V B-operand is K-contiguous)
__global__ __launch_bounds__(256) void vtrans_kernel(const unsigned short* __restrict__ src,
                                                     unsigned short* __restrict__ dst) {
  int idx = blockIdx.x * blockDim.x + threadIdx.x;
  if (idx >= MM * CC) return;
  int d = idx % DD; int r = idx / DD;
  int h = r % HH;   r /= HH;
  int t = r % TT;   int b = r / TT;
  dst[((size_t)(b * HH + h) * DD + d) * TT + t] = src[idx];
}

// ---------------- WMMA GEMM: out[m][n] = sum_k A[m][k] * W[n][k] ------------
// Each wave: 32 rows x 64 cols (2 A frags x 4 B frags -> 8 WMMAs per K-step,
// 1.5 b128 loads per WMMA). Block = 8 waves = 256 rows x 64 cols.
template <bool F32OUT>
__global__ __launch_bounds__(256) void gemm_wmma(const unsigned short* __restrict__ A,
                                                 const unsigned short* __restrict__ W,
                                                 void* __restrict__ out,
                                                 int M, int N, int K) {
  const int w = threadIdx.x >> 5, lane = threadIdx.x & 31;
  const int row0 = (blockIdx.x * 8 + w) * 32;
  const int col0 = blockIdx.y * 64;
  if (row0 >= M) return;

  f32x8 acc[2][4] = {};
  const unsigned short* A0 = A + (size_t)row0 * K;
  const unsigned short* A1 = A + (size_t)(row0 + 16) * K;
  for (int k0 = 0; k0 < K; k0 += 32) {
    f16x16 a0 = load_frag_A(A0 + k0, K, lane);
    f16x16 a1 = load_frag_A(A1 + k0, K, lane);
#pragma unroll
    for (int n = 0; n < 4; ++n) {
      f16x16 b = load_frag_B(W + (size_t)(col0 + 16 * n) * K + k0, K, lane);
      acc[0][n] = wmma_f16(a0, b, acc[0][n]);
      acc[1][n] = wmma_f16(a1, b, acc[1][n]);
    }
  }
  const int half = lane >> 4, ln = lane & 15;
#pragma unroll
  for (int g = 0; g < 2; ++g)
#pragma unroll
    for (int n = 0; n < 4; ++n)
#pragma unroll
      for (int r = 0; r < 8; ++r) {
        size_t off = (size_t)(row0 + g * 16 + r + 8 * half) * N + (col0 + n * 16 + ln);
        float val = acc[g][n][r];
        if (F32OUT) ((float*)out)[off] = val;
        else        ((unsigned short*)out)[off] = f2h(val);
      }
}

// ---------------- causal flash attention ------------------------------------
// One wave handles a 16-row query tile for one (b,h). O (16x64) accumulated in f32.
__global__ __launch_bounds__(256) void flash_attn_kernel(const unsigned short* __restrict__ qh,
                                                         const unsigned short* __restrict__ kh,
                                                         const unsigned short* __restrict__ vt,
                                                         unsigned short* __restrict__ attnout) {
  __shared__ __align__(16) unsigned short pshm[8][16 * 32];
  const int w = threadIdx.x >> 5, lane = threadIdx.x & 31;
  const int half = lane >> 4, ln = lane & 15;
  const int bh = blockIdx.y;
  const int qbase = (blockIdx.x * 8 + w) * 16;

  const unsigned short* Q  = qh + (size_t)bh * TT * DD + (size_t)qbase * DD;
  const unsigned short* Kp = kh + (size_t)bh * TT * DD;
  const unsigned short* Vp = vt + (size_t)bh * DD * TT;

  f16x16 qa0 = load_frag_A(Q,      DD, lane);   // d = 0..31
  f16x16 qa1 = load_frag_A(Q + 32, DD, lane);   // d = 32..63

  f32x8 o[4] = {};
  float mrow[8], lrow[8];
#pragma unroll
  for (int r = 0; r < 8; ++r) { mrow[r] = -3.0e38f; lrow[r] = 0.0f; }

  const int nchunk = qbase / 32 + 1;            // chunks of 32 keys up to diagonal
  for (int jc = 0; jc < nchunk; ++jc) {
    const int j0 = jc * 32;
    // S = Q*K^T for 16 rows x 32 cols (two 16x16 tiles, K=64 via 2 WMMAs each)
    f32x8 s0 = {}, s1 = {};
    {
      f16x16 kb0 = load_frag_B(Kp + (size_t)j0 * DD,      DD, lane);
      f16x16 kb1 = load_frag_B(Kp + (size_t)j0 * DD + 32, DD, lane);
      s0 = wmma_f16(qa0, kb0, s0);
      s0 = wmma_f16(qa1, kb1, s0);
    }
    {
      f16x16 kb0 = load_frag_B(Kp + (size_t)(j0 + 16) * DD,      DD, lane);
      f16x16 kb1 = load_frag_B(Kp + (size_t)(j0 + 16) * DD + 32, DD, lane);
      s1 = wmma_f16(qa0, kb0, s1);
      s1 = wmma_f16(qa1, kb1, s1);
    }
    if (jc == nchunk - 1) {                     // diagonal chunk: causal mask
#pragma unroll
      for (int r = 0; r < 8; ++r) {
        int i = qbase + r + 8 * half;
        if (j0 + ln > i)      s0[r] = -3.0e38f;
        if (j0 + 16 + ln > i) s1[r] = -3.0e38f;
      }
    }
    // row max across the 16 lanes holding each row (lanes 0-15 rows 0-7, 16-31 rows 8-15)
    float rmax[8];
#pragma unroll
    for (int r = 0; r < 8; ++r) rmax[r] = fmaxf(s0[r], s1[r]);
#pragma unroll
    for (int mk = 1; mk <= 8; mk <<= 1)
#pragma unroll
      for (int r = 0; r < 8; ++r)
        rmax[r] = fmaxf(rmax[r], __shfl_xor(rmax[r], mk, 32));

    float rs[8];
#pragma unroll
    for (int r = 0; r < 8; ++r) {
      float mnew = fmaxf(mrow[r], rmax[r]);
      float corr = __expf(mrow[r] - mnew);
      mrow[r] = mnew;
      lrow[r] *= corr;
#pragma unroll
      for (int n = 0; n < 4; ++n) o[n][r] *= corr;
      float p0 = __expf(s0[r] - mnew);
      float p1 = __expf(s1[r] - mnew);
      rs[r] = p0 + p1;
      // stage P tile in LDS (row-major 16x32) to re-load in A-layout
      pshm[w][(r + 8 * half) * 32 + ln]      = f2h(p0);
      pshm[w][(r + 8 * half) * 32 + 16 + ln] = f2h(p1);
    }
#pragma unroll
    for (int mk = 1; mk <= 8; mk <<= 1)
#pragma unroll
      for (int r = 0; r < 8; ++r)
        rs[r] += __shfl_xor(rs[r], mk, 32);
#pragma unroll
    for (int r = 0; r < 8; ++r) lrow[r] += rs[r];

    asm volatile("s_wait_dscnt 0" ::: "memory");  // LDS store -> load, same wave
    f16x16 pa = load_frag_A(&pshm[w][0], 32, lane);
#pragma unroll
    for (int n = 0; n < 4; ++n) {
      // B[k=j][n=d] from Vt[d][t]: contiguous in j per lane
      f16x16 vb = load_frag_B(Vp + (size_t)(n * 16) * TT + j0, TT, lane);
      o[n] = wmma_f16(pa, vb, o[n]);
    }
  }

  const int b = bh / HH, h = bh % HH;
  unsigned short* outp = attnout + (size_t)(b * TT + qbase) * CC + h * DD;
#pragma unroll
  for (int r = 0; r < 8; ++r) {
    float inv = 1.0f / lrow[r];
#pragma unroll
    for (int n = 0; n < 4; ++n)
      outp[(size_t)(r + 8 * half) * CC + n * 16 + ln] = f2h(o[n][r] * inv);
  }
}

// ---------------- host orchestration ----------------------------------------
extern "C" void kernel_launch(void* const* d_in, const int* in_sizes, int n_in,
                              void* d_out, int out_size, void* d_ws, size_t ws_size,
                              hipStream_t stream) {
  (void)in_sizes; (void)n_in; (void)out_size; (void)ws_size;
  const float* x  = (const float*)d_in[0];
  const float* Wq = (const float*)d_in[1];
  const float* Wk = (const float*)d_in[2];
  const float* Wv = (const float*)d_in[3];
  const float* Wo = (const float*)d_in[4];

  unsigned char* p = (unsigned char*)d_ws;
  auto take = [&](size_t elems) {
    unsigned short* r = (unsigned short*)p;
    p += (elems * sizeof(unsigned short) + 255) & ~(size_t)255;
    return r;
  };
  unsigned short* xb   = take((size_t)MM * CC);
  unsigned short* wqb  = take((size_t)CC * CC);
  unsigned short* wkb  = take((size_t)CC * CC);
  unsigned short* wvb  = take((size_t)CC * CC);
  unsigned short* wob  = take((size_t)CC * CC);
  unsigned short* qraw = take((size_t)MM * CC);
  unsigned short* kraw = take((size_t)MM * CC);
  unsigned short* vraw = take((size_t)MM * CC);
  unsigned short* qh   = take((size_t)MM * CC);
  unsigned short* khh  = take((size_t)MM * CC);
  unsigned short* vt   = take((size_t)MM * CC);
  unsigned short* ao   = take((size_t)MM * CC);

  const int thr = 256;
  f32_to_f16_kernel<<<(MM * CC + thr - 1) / thr, thr, 0, stream>>>(x,  xb,  MM * CC);
  f32_to_f16_kernel<<<(CC * CC + thr - 1) / thr, thr, 0, stream>>>(Wq, wqb, CC * CC);
  f32_to_f16_kernel<<<(CC * CC + thr - 1) / thr, thr, 0, stream>>>(Wk, wkb, CC * CC);
  f32_to_f16_kernel<<<(CC * CC + thr - 1) / thr, thr, 0, stream>>>(Wv, wvb, CC * CC);
  f32_to_f16_kernel<<<(CC * CC + thr - 1) / thr, thr, 0, stream>>>(Wo, wob, CC * CC);

  dim3 gg(MM / 256, CC / 64);                   // 32 rows/wave * 8 waves = 256 rows/block
  gemm_wmma<false><<<gg, thr, 0, stream>>>(xb, wqb, qraw, MM, CC, CC);
  gemm_wmma<false><<<gg, thr, 0, stream>>>(xb, wkb, kraw, MM, CC, CC);
  gemm_wmma<false><<<gg, thr, 0, stream>>>(xb, wvb, vraw, MM, CC, CC);

  const int ropeN = BB * TT * HH * (DD / 2);
  rope_split_kernel<<<(ropeN + thr - 1) / thr, thr, 0, stream>>>(qraw, qh, 0.125f); // 1/sqrt(64)
  rope_split_kernel<<<(ropeN + thr - 1) / thr, thr, 0, stream>>>(kraw, khh, 1.0f);
  vtrans_kernel<<<(MM * CC + thr - 1) / thr, thr, 0, stream>>>(vraw, vt);

  flash_attn_kernel<<<dim3(TT / 16 / 8, BB * HH), thr, 0, stream>>>(qh, khh, vt, ao);

  gemm_wmma<true><<<gg, thr, 0, stream>>>(ao, wob, d_out, MM, CC, CC);
}
